// SupportTokenizer_62156766707857
// MI455X (gfx1250) — matrix operand order repaired and checked by reading, not compiled, not run on hardware
//
#include <hip/hip_runtime.h>
#include <hip/hip_bf16.h>

typedef __attribute__((ext_vector_type(16))) _Float16 v16h;
typedef __attribute__((ext_vector_type(8)))  _Float16 v8h;
typedef __attribute__((ext_vector_type(8)))  float    v8f;

#define N_B   64
#define C_CH  160
#define LTOK  4096
#define DDIM  160
#define MSLOT 6
#define EPS_LN 1e-5f

static __device__ __forceinline__ v8f wmma16(v16h a, v16h b, v8f c) {
  // D = A(16x32 f16) * B(32x16 f16) + C(16x16 f32)
  return __builtin_amdgcn_wmma_f32_16x16x32_f16(false, a, false, b, (short)0, c, false, false);
}

// A-fragment K index for half j of lane (contiguous K order: K = grp*16 + j)
static __device__ __forceinline__ int a_kidx(int j, int grp) {
  return j + grp * 16;
}

// ---------------------------------------------------------------------------
// 0) Pack Wk / Wv (f32 [C,D]) into f16 B-fragment order:
//    dst[((jtile*5 + s)*32 + lane)*16 + h]  with K = s*32 + h + (lane>>4)*16,
//    N = jtile*16 + (lane&15)
// ---------------------------------------------------------------------------
__global__ void pack_weights_kernel(const float* __restrict__ Wk,
                                    const float* __restrict__ Wv,
                                    _Float16* __restrict__ wk_pack,
                                    _Float16* __restrict__ wv_pack) {
  int idx = blockIdx.x * 256 + threadIdx.x;         // 2 * 25600
  if (idx >= 2 * 25600) return;
  int which = idx / 25600;
  int r = idx % 25600;
  int h    = r & 15;
  int lane = (r >> 4) & 31;
  int s    = (r >> 9) % 5;
  int j    = r / 2560;
  int K  = s * 32 + h + (lane >> 4) * 16;
  int dd = j * 16 + (lane & 15);
  const float* W = which ? Wv : Wk;
  _Float16* dst  = which ? wv_pack : wk_pack;
  dst[r] = (_Float16)W[K * DDIM + dd];
}

// ---------------------------------------------------------------------------
// 0b) Broadcast slots_init -> slots[n][m][d]
// ---------------------------------------------------------------------------
__global__ void init_slots_kernel(const float* __restrict__ slots_init,
                                  float* __restrict__ slots) {
  int idx = blockIdx.x * 256 + threadIdx.x;         // N*M*D = 61440
  if (idx >= N_B * MSLOT * DDIM) return;
  int d = idx % DDIM;
  int m = (idx / DDIM) % MSLOT;
  slots[idx] = slots_init[m * DDIM + d];
}

// ---------------------------------------------------------------------------
// 1) Fused LN + (k = x@Wk, v = x@Wv) via WMMA.  One wave handles 16 tokens.
//    k stored [n][l][d] f16;  v stored transposed [n][d][l] f16 via packed
//    b128 stores (each lane's 8 accumulator rows are 8 consecutive tokens).
// ---------------------------------------------------------------------------
__global__ __launch_bounds__(32)
void kv_kernel(const float* __restrict__ feat,
               const float* __restrict__ kv_g,
               const float* __restrict__ kv_b,
               const _Float16* __restrict__ wk_pack,
               const _Float16* __restrict__ wv_pack,
               _Float16* __restrict__ k_h,
               _Float16* __restrict__ v_t) {
  __shared__ float x[16][C_CH];          // token-major f32 tile
  __shared__ float gsh[C_CH], bsh[C_CH];
  __shared__ float mu[16], rs[16];
  __shared__ _Float16 apack[5][32][16];  // A-fragments, 5 K-steps

  const int lane = threadIdx.x;
  const int tile = blockIdx.x;                   // N*L/16 tiles
  const int n  = tile >> 8;                      // 256 tiles per batch
  const int l0 = (tile & 255) << 4;

  for (int i = lane; i < C_CH; i += 32) { gsh[i] = kv_g[i]; bsh[i] = kv_b[i]; }
  // transpose-load feat[n][c][l0..l0+15]; lanes 0..15 -> consecutive tokens
  for (int i = lane; i < 16 * C_CH; i += 32) {
    int t = i & 15, c = i >> 4;
    x[t][c] = feat[(size_t)n * C_CH * LTOK + (size_t)c * LTOK + l0 + t];
  }
  __syncthreads();

  if (lane < 16) {
    float s = 0.f, s2 = 0.f;
    for (int c = 0; c < C_CH; c++) { float v = x[lane][c]; s += v; s2 += v * v; }
    float m = s * (1.0f / C_CH);
    mu[lane] = m;
    rs[lane] = rsqrtf(fmaxf(s2 * (1.0f / C_CH) - m * m, 0.f) + EPS_LN);
  }
  __syncthreads();

  const int m16 = lane & 15, grp = lane >> 4;
  {
    float mm = mu[m16], rr = rs[m16];
#pragma unroll
    for (int s = 0; s < 5; s++)
#pragma unroll
      for (int j = 0; j < 16; j++) {
        int c = s * 32 + a_kidx(j, grp);
        float val = (x[m16][c] - mm) * rr * gsh[c] + bsh[c];
        apack[s][lane][j] = (_Float16)val;
      }
  }
  __syncthreads();

  v16h A[5];
#pragma unroll
  for (int s = 0; s < 5; s++) A[s] = *(const v16h*)(&apack[s][lane][0]);

  for (int j = 0; j < 10; j++) {
    v8f ak = {}, av = {};
#pragma unroll
    for (int s = 0; s < 5; s++) {
      v16h bk = *(const v16h*)(wk_pack + ((j * 5 + s) * 32 + lane) * 16);
      v16h bv = *(const v16h*)(wv_pack + ((j * 5 + s) * 32 + lane) * 16);
      ak = wmma16(A[s], bk, ak);
      av = wmma16(A[s], bv, av);
    }
    int dd = j * 16 + m16;
    // k: lanes 0..15 write token r's contiguous 32B row segment (coalesced)
#pragma unroll
    for (int r = 0; r < 8; r++) {
      int tok = r + grp * 8;
      k_h[((size_t)n * LTOK + l0 + tok) * DDIM + dd] = (_Float16)ak[r];
    }
    // v (transposed): lane's 8 rows are 8 consecutive tokens at fixed dd ->
    // one contiguous 16B vector store per lane per dim tile
    v8h vh;
#pragma unroll
    for (int r = 0; r < 8; r++) vh[r] = (_Float16)av[r];
    *(v8h*)(v_t + ((size_t)n * DDIM + dd) * LTOK + l0 + grp * 8) = vh;
  }
}

// ---------------------------------------------------------------------------
// 2) q = LN(slots) @ Wq * (scale/tau), emitted as packed A-fragments
// ---------------------------------------------------------------------------
__global__ __launch_bounds__(160)
void q_kernel(const float* __restrict__ slots,
              const float* __restrict__ q_g,
              const float* __restrict__ q_b,
              const float* __restrict__ Wq,
              const float* __restrict__ log_tau,
              _Float16* __restrict__ q_pack) {
  __shared__ float sl[MSLOT][DDIM];
  __shared__ float ln[MSLOT][DDIM];
  __shared__ float qo[MSLOT][DDIM];
  __shared__ float mu[MSLOT], rs[MSLOT];
  const int n = blockIdx.x, tid = threadIdx.x;

  for (int i = tid; i < MSLOT * DDIM; i += 160) {
    int m = i / DDIM, c = i % DDIM;
    sl[m][c] = slots[(n * MSLOT + m) * DDIM + c];
  }
  __syncthreads();
  if (tid < MSLOT) {
    float s = 0.f, s2 = 0.f;
    for (int c = 0; c < DDIM; c++) { float v = sl[tid][c]; s += v; s2 += v * v; }
    float m = s * (1.0f / DDIM);
    mu[tid] = m;
    rs[tid] = rsqrtf(fmaxf(s2 * (1.0f / DDIM) - m * m, 0.f) + EPS_LN);
  }
  __syncthreads();
  for (int i = tid; i < MSLOT * DDIM; i += 160) {
    int m = i / DDIM, c = i % DDIM;
    ln[m][c] = (sl[m][c] - mu[m]) * rs[m] * q_g[c] + q_b[c];
  }
  __syncthreads();

  float tau  = log1pf(expf(log_tau[0])) + 0.5f;      // softplus + 0.5
  float coef = rsqrtf((float)DDIM) / tau;            // scale / tau folded into q
  {
    float acc[MSLOT];
#pragma unroll
    for (int m = 0; m < MSLOT; m++) acc[m] = 0.f;
    for (int c = 0; c < DDIM; c++) {
      float w = Wq[c * DDIM + tid];
#pragma unroll
      for (int m = 0; m < MSLOT; m++) acc[m] += ln[m][c] * w;
    }
#pragma unroll
    for (int m = 0; m < MSLOT; m++) qo[m][tid] = acc[m] * coef;
  }
  __syncthreads();

  // emit packed A-fragments (rows 6..15 zero)
  for (int u = 0; u < 16; u++) {
    int idx  = tid * 16 + u;                 // 0..2559
    int jj   = idx & 15;
    int lane = (idx >> 4) & 31;
    int s    = idx >> 9;
    int m = lane & 15;
    int c = s * 32 + a_kidx(jj, lane >> 4);
    q_pack[(size_t)n * 2560 + idx] = (m < MSLOT) ? (_Float16)qo[m][c] : (_Float16)0.f;
  }
}

// ---------------------------------------------------------------------------
// 3) logits = q @ k^T via WMMA, softmax over slots (M=6) per token.
//    Each wave: one 16-token tile. C-tile layout puts m=0..5 for a token in
//    acc[0..5] of lanes 0..15.
// ---------------------------------------------------------------------------
__global__ __launch_bounds__(128)
void attn_kernel(const _Float16* __restrict__ q_pack,
                 const _Float16* __restrict__ k_h,
                 float* __restrict__ attn) {
  const int wave = threadIdx.x >> 5, lane = threadIdx.x & 31;
  const int tile = blockIdx.x * 4 + wave;        // N*L/16 tiles total
  const int n  = tile >> 8;
  const int l0 = (tile & 255) << 4;

  v8f acc = {};
#pragma unroll
  for (int s = 0; s < 5; s++) {
    v16h a = *(const v16h*)(q_pack + ((size_t)n * 5 + s) * 512 + lane * 16);
    const _Float16* bp = k_h + ((size_t)n * LTOK + l0 + (lane & 15)) * DDIM
                             + s * 32 + (lane >> 4) * 16;
    v16h b = *(const v16h*)bp;
    acc = wmma16(a, b, acc);
  }
  if (lane < 16) {
    int l = l0 + lane;
    float mx = acc[0];
#pragma unroll
    for (int r = 1; r < MSLOT; r++) mx = fmaxf(mx, acc[r]);
    float e[MSLOT], sum = 0.f;
#pragma unroll
    for (int r = 0; r < MSLOT; r++) { e[r] = expf(acc[r] - mx); sum += e[r]; }
    float inv = 1.0f / sum;
#pragma unroll
    for (int r = 0; r < MSLOT; r++)
      attn[((size_t)n * MSLOT + r) * LTOK + l] = e[r] * inv;
  }
}

// ---------------------------------------------------------------------------
// 4) attn_sum[n][m] = max(sum_l attn, 1e-6)   (block reduction, deterministic)
// ---------------------------------------------------------------------------
__global__ __launch_bounds__(256)
void attn_sum_kernel(const float* __restrict__ attn,
                     float* __restrict__ attn_sum) {
  __shared__ float red[256];
  const int nm = blockIdx.x, tid = threadIdx.x;
  float s = 0.f;
  for (int l = tid; l < LTOK; l += 256) s += attn[(size_t)nm * LTOK + l];
  red[tid] = s;
  __syncthreads();
  for (int off = 128; off > 0; off >>= 1) {
    if (tid < off) red[tid] += red[tid + off];
    __syncthreads();
  }
  if (tid == 0) attn_sum[nm] = fmaxf(red[0], 1e-6f);
}

// ---------------------------------------------------------------------------
// 5) updates = (attn @ v) / attn_sum   via WMMA; 10 waves = 10 dim tiles
// ---------------------------------------------------------------------------
__global__ __launch_bounds__(320)
void updates_kernel(const float* __restrict__ attn,
                    const float* __restrict__ attn_sum,
                    const _Float16* __restrict__ v_t,
                    float* __restrict__ upd) {
  __shared__ _Float16 apack[4][32][16];     // A-fragments for a 128-token chunk
  const int n = blockIdx.x;
  const int tid = threadIdx.x;
  const int wave = tid >> 5, lane = tid & 31;
  const int jt = wave;                      // dim tile 0..9

  v8f acc = {};
  for (int t0 = 0; t0 < LTOK; t0 += 128) {
    __syncthreads();
    if (tid < 256) {
#pragma unroll
      for (int u = 0; u < 8; u++) {
        int idx  = tid * 8 + u;             // 0..2047
        int jj   = idx & 15;
        int ln2  = (idx >> 4) & 31;
        int ss   = idx >> 9;
        int m    = ln2 & 15;
        int tok  = t0 + ss * 32 + a_kidx(jj, ln2 >> 4);
        float v  = (m < MSLOT) ? attn[((size_t)n * MSLOT + m) * LTOK + tok] : 0.f;
        apack[ss][ln2][jj] = (_Float16)v;
      }
    }
    __syncthreads();
#pragma unroll
    for (int ss = 0; ss < 4; ss++) {
      v16h a = *(const v16h*)(&apack[ss][lane][0]);
      const _Float16* bp = v_t + ((size_t)n * DDIM + jt * 16 + (lane & 15)) * LTOK
                               + t0 + ss * 32 + (lane >> 4) * 16;
      v16h b = *(const v16h*)bp;
      acc = wmma16(a, b, acc);
    }
  }
#pragma unroll
  for (int r = 0; r < 8; r++) {
    int m = r + (lane >> 4) * 8;
    if (m < MSLOT) {
      int d = jt * 16 + (lane & 15);
      upd[(n * MSLOT + m) * DDIM + d] = acc[r] / attn_sum[n * MSLOT + m];
    }
  }
}

// ---------------------------------------------------------------------------
// 6) slots = LN(slots+upd); slots += GELU(slots@W1+b1)@W2+b2   (f32 VALU)
// ---------------------------------------------------------------------------
__global__ __launch_bounds__(160)
void slot_update_kernel(float* __restrict__ slots,
                        const float* __restrict__ upd,
                        const float* __restrict__ out_g,
                        const float* __restrict__ out_b,
                        const float* __restrict__ W1,
                        const float* __restrict__ b1,
                        const float* __restrict__ W2,
                        const float* __restrict__ b2) {
  __shared__ float s0[MSLOT][DDIM];
  __shared__ float sln[MSLOT][DDIM];
  __shared__ float hb[MSLOT][2 * DDIM];
  __shared__ float mu[MSLOT], rs[MSLOT];
  const int n = blockIdx.x, tid = threadIdx.x;

  for (int i = tid; i < MSLOT * DDIM; i += 160) {
    int m = i / DDIM, c = i % DDIM;
    s0[m][c] = slots[(n * MSLOT + m) * DDIM + c] + upd[(n * MSLOT + m) * DDIM + c];
  }
  __syncthreads();
  if (tid < MSLOT) {
    float s = 0.f, s2 = 0.f;
    for (int c = 0; c < DDIM; c++) { float v = s0[tid][c]; s += v; s2 += v * v; }
    float m = s * (1.0f / DDIM);
    mu[tid] = m;
    rs[tid] = rsqrtf(fmaxf(s2 * (1.0f / DDIM) - m * m, 0.f) + EPS_LN);
  }
  __syncthreads();
  for (int i = tid; i < MSLOT * DDIM; i += 160) {
    int m = i / DDIM, c = i % DDIM;
    sln[m][c] = (s0[m][c] - mu[m]) * rs[m] * out_g[c] + out_b[c];
  }
  __syncthreads();
  for (int h = tid; h < 2 * DDIM; h += 160) {
    float acc[MSLOT];
#pragma unroll
    for (int m = 0; m < MSLOT; m++) acc[m] = b1[h];
    for (int c = 0; c < DDIM; c++) {
      float w = W1[c * (2 * DDIM) + h];
#pragma unroll
      for (int m = 0; m < MSLOT; m++) acc[m] += sln[m][c] * w;
    }
#pragma unroll
    for (int m = 0; m < MSLOT; m++) {
      float x = acc[m];
      hb[m][h] = 0.5f * x * (1.0f + erff(x * 0.70710678118654752f));  // exact GELU
    }
  }
  __syncthreads();
  {
    int d = tid;
    float acc[MSLOT];
#pragma unroll
    for (int m = 0; m < MSLOT; m++) acc[m] = b2[d];
    for (int h = 0; h < 2 * DDIM; h++) {
      float w = W2[h * DDIM + d];
#pragma unroll
      for (int m = 0; m < MSLOT; m++) acc[m] += hb[m][h] * w;
    }
#pragma unroll
    for (int m = 0; m < MSLOT; m++)
      slots[(n * MSLOT + m) * DDIM + d] = sln[m][d] + acc[m];
  }
}

// ---------------------------------------------------------------------------
// 7) d_out = [slots (N*M*D)] ++ [attn / clip(sum) (N*M*L)]
// ---------------------------------------------------------------------------
__global__ void finalize_kernel(const float* __restrict__ slots,
                                const float* __restrict__ attn,
                                const float* __restrict__ attn_sum,
                                float* __restrict__ out) {
  const int TS = N_B * MSLOT * DDIM;
  const size_t TA = (size_t)N_B * MSLOT * LTOK;
  size_t idx = (size_t)blockIdx.x * 256 + threadIdx.x;
  if (idx < (size_t)TS) {
    out[idx] = slots[idx];
  } else if (idx < TS + TA) {
    size_t i = idx - TS;
    size_t nm = i / LTOK;
    out[idx] = attn[i] / attn_sum[nm];
  }
}

// ---------------------------------------------------------------------------
extern "C" void kernel_launch(void* const* d_in, const int* in_sizes, int n_in,
                              void* d_out, int out_size, void* d_ws, size_t ws_size,
                              hipStream_t stream) {
  (void)in_sizes; (void)n_in; (void)out_size; (void)ws_size;
  const float* feat       = (const float*)d_in[0];
  const float* slots_init = (const float*)d_in[1];
  const float* log_tau    = (const float*)d_in[2];
  const float* kv_g       = (const float*)d_in[3];
  const float* kv_b       = (const float*)d_in[4];
  const float* Wk         = (const float*)d_in[5];
  const float* Wv         = (const float*)d_in[6];
  const float* q_g        = (const float*)d_in[7];
  const float* q_b        = (const float*)d_in[8];
  const float* Wq         = (const float*)d_in[9];
  const float* out_g      = (const float*)d_in[10];
  const float* out_b      = (const float*)d_in[11];
  const float* W1         = (const float*)d_in[12];
  const float* b1         = (const float*)d_in[13];
  const float* W2         = (const float*)d_in[14];
  const float* b2         = (const float*)d_in[15];
  float* out = (float*)d_out;

  char* w = (char*)d_ws;
  size_t o = 0;
  _Float16* wk_pack = (_Float16*)(w + o); o += 25600 * 2;
  _Float16* wv_pack = (_Float16*)(w + o); o += 25600 * 2;
  _Float16* k_h     = (_Float16*)(w + o); o += (size_t)N_B * LTOK * DDIM * 2;
  _Float16* v_t     = (_Float16*)(w + o); o += (size_t)N_B * LTOK * DDIM * 2;
  _Float16* q_pack  = (_Float16*)(w + o); o += (size_t)N_B * 2560 * 2;
  float* attn       = (float*)(w + o);    o += (size_t)N_B * MSLOT * LTOK * 4;
  float* attn_sum   = (float*)(w + o);    o += N_B * MSLOT * 4;
  float* slots      = (float*)(w + o);    o += N_B * MSLOT * DDIM * 4;
  float* upd        = (float*)(w + o);    o += N_B * MSLOT * DDIM * 4;

  pack_weights_kernel<<<(2 * 25600 + 255) / 256, 256, 0, stream>>>(Wk, Wv, wk_pack, wv_pack);
  init_slots_kernel<<<(N_B * MSLOT * DDIM + 255) / 256, 256, 0, stream>>>(slots_init, slots);
  kv_kernel<<<N_B * LTOK / 16, 32, 0, stream>>>(feat, kv_g, kv_b, wk_pack, wv_pack, k_h, v_t);

  for (int it = 0; it < 2; it++) {
    q_kernel<<<N_B, 160, 0, stream>>>(slots, q_g, q_b, Wq, log_tau, q_pack);
    attn_kernel<<<N_B * LTOK / 64, 128, 0, stream>>>(q_pack, k_h, attn);
    attn_sum_kernel<<<N_B * MSLOT, 256, 0, stream>>>(attn, attn_sum);
    updates_kernel<<<N_B, 320, 0, stream>>>(attn, attn_sum, v_t, upd);
    slot_update_kernel<<<N_B, 160, 0, stream>>>(slots, upd, out_g, out_b, W1, b1, W2, b2);
  }

  size_t total = (size_t)N_B * MSLOT * DDIM + (size_t)N_B * MSLOT * LTOK;
  finalize_kernel<<<(unsigned)((total + 255) / 256), 256, 0, stream>>>(slots, attn, attn_sum, out);
}
